// Qwen3Attention_52639119179931
// MI455X (gfx1250) — compile-verified, matrix operand order checked by reading, not run on hardware
//
#include <hip/hip_runtime.h>

typedef __attribute__((ext_vector_type(16))) __bf16 v16bf;
typedef __attribute__((ext_vector_type(8)))  float  v8f;

#define T_SEQ 4096
#define HID   1024
#define NQH   16
#define NKVH  8
#define DHEAD 128

union BFrag {
    v16bf v;
    unsigned int u[8];
    unsigned short h[16];
};

__device__ __forceinline__ unsigned short f2bf(float f) {
    unsigned int u = __float_as_uint(f);
    u = (u + 0x7FFFu + ((u >> 16) & 1u)) >> 16;
    return (unsigned short)u;
}

// k-pair offset pattern for 16-bit A/B fragments (16x32, wave32):
// lanes 0-15 cover K={0..7,16..23}, lanes 16-31 cover K={8..15,24..31}
__device__ __forceinline__ int kpair(int i, int kb8) {
    return kb8 + ((i < 4) ? (2 * i) : (16 + 2 * (i - 4)));
}

// Async copy 32B global bf16 -> LDS (CDNA5 ASYNCcnt path). The 24-bit inst
// offset applies to BOTH the global and the LDS address.
__device__ __forceinline__ void async_cp32B(const unsigned short* g, unsigned short* l) {
    unsigned lds_off = (unsigned)(size_t)l;
    asm volatile("global_load_async_to_lds_b128 %0, %1, off"
                 :: "v"(lds_off), "v"(g) : "memory");
    asm volatile("global_load_async_to_lds_b128 %0, %1, off offset:16"
                 :: "v"(lds_off), "v"(g) : "memory");
}
__device__ __forceinline__ void wait_async0() {
    asm volatile("s_wait_asynccnt 0" ::: "memory");
}
// previous tile's 4 ops complete once outstanding <= 4 (async loads finish in order)
__device__ __forceinline__ void wait_async4() {
    asm volatile("s_wait_asynccnt 4" ::: "memory");
}

// ---------------------------------------------------------------------------
// fp32 -> bf16 elementwise (4 per thread)
// ---------------------------------------------------------------------------
__global__ __launch_bounds__(256)
void cvt_bf16_kernel(const float* __restrict__ in, unsigned short* __restrict__ out) {
    int i = (blockIdx.x * 256 + threadIdx.x) * 4;
    float4 f = *(const float4*)(in + i);
    out[i + 0] = f2bf(f.x);
    out[i + 1] = f2bf(f.y);
    out[i + 2] = f2bf(f.z);
    out[i + 3] = f2bf(f.w);
}

// ---------------------------------------------------------------------------
// fp32 [K][N] -> bf16 transposed [N][K]  (one-time weight prep)
// ---------------------------------------------------------------------------
__global__ __launch_bounds__(256)
void cvt_bf16_t_kernel(const float* __restrict__ in, unsigned short* __restrict__ out,
                       int K, int N) {
    int gid = blockIdx.x * 256 + threadIdx.x;
    int k = gid / N;
    int n = gid - k * N;
    out[(size_t)n * K + k] = f2bf(in[gid]);
}

// ---------------------------------------------------------------------------
// C[M,N](f32) = A[M,K](bf16, row-major) @ Bt[N,K](bf16, pre-transposed).
// Block 128x128, K-step 32, 256 threads = 8 waves (4x2 wave grid).
// Both tiles double-buffered via async global->LDS copies.
// ---------------------------------------------------------------------------
__global__ __launch_bounds__(256)
void gemm_bf16_128(const unsigned short* __restrict__ A,
                   const unsigned short* __restrict__ Bt,
                   float* __restrict__ C, int M, int N, int K) {
    __shared__ unsigned short a_lds[2][128 * 32];   // [m][k]
    __shared__ unsigned short b_lds[2][128 * 32];   // [n][k]

    const int n0 = blockIdx.x * 128;
    const int m0 = blockIdx.y * 128;
    const int tid  = threadIdx.x;
    const int wave = tid >> 5;
    const int lane = tid & 31;
    const int wm = wave >> 1;          // 0..3
    const int wn = wave & 1;           // 0..1
    const int ln = lane & 15;
    const int kb8 = (lane >> 4) * 8;

    v8f acc[2][4];
#pragma unroll
    for (int mt = 0; mt < 2; ++mt)
#pragma unroll
        for (int nt = 0; nt < 4; ++nt) acc[mt][nt] = (v8f)0.0f;

    const int sr = tid >> 1;                 // 0..127 stage row (A and B tiles)
    const int sc = (tid & 1) * 16;           // 0/16   stage col base (halfs)
    const unsigned short* gA = A  + (size_t)(m0 + sr) * K + sc;
    const unsigned short* gB = Bt + (size_t)(n0 + sr) * K + sc;

    const int ntiles = K >> 5;
    // prologue: stage tile 0 into buffer 0
    async_cp32B(gA, &a_lds[0][sr * 32 + sc]);
    async_cp32B(gB, &b_lds[0][sr * 32 + sc]);

    for (int it = 0; it < ntiles; ++it) {
        const int cur = it & 1;
        if (it + 1 < ntiles) {
            __syncthreads();   // next buffer no longer being read
            const int k0n = (it + 1) << 5;
            async_cp32B(gA + k0n, &a_lds[cur ^ 1][sr * 32 + sc]);
            async_cp32B(gB + k0n, &b_lds[cur ^ 1][sr * 32 + sc]);
            wait_async4();     // tile `it` (4 older ops) now resident
        } else {
            wait_async0();
        }
        __syncthreads();       // all waves' async writes for tile `it` visible

        const unsigned short* ab = a_lds[cur];
        const unsigned short* bb = b_lds[cur];
        BFrag af[2];
#pragma unroll
        for (int mt = 0; mt < 2; ++mt) {
            int r = (wm * 32 + mt * 16 + ln) * 32;
#pragma unroll
            for (int i = 0; i < 8; ++i)
                af[mt].u[i] = *(const unsigned int*)&ab[r + kpair(i, kb8)];
        }
#pragma unroll
        for (int nt = 0; nt < 4; ++nt) {
            BFrag bf_;
            int c = (wn * 64 + nt * 16 + ln) * 32;
#pragma unroll
            for (int i = 0; i < 8; ++i)
                bf_.u[i] = *(const unsigned int*)&bb[c + kpair(i, kb8)];
#pragma unroll
            for (int mt = 0; mt < 2; ++mt)
                acc[mt][nt] = __builtin_amdgcn_wmma_f32_16x16x32_bf16(
                    false, af[mt].v, false, bf_.v, (short)0, acc[mt][nt], false, false);
        }
    }

#pragma unroll
    for (int mt = 0; mt < 2; ++mt)
#pragma unroll
        for (int nt = 0; nt < 4; ++nt)
#pragma unroll
            for (int vp = 0; vp < 8; ++vp) {
                int row = m0 + wm * 32 + mt * 16 + vp + kb8;
                int col = n0 + wn * 64 + nt * 16 + ln;
                C[(size_t)row * N + col] = acc[mt][nt][vp];
            }
}

// ---------------------------------------------------------------------------
// RMSNorm + rotary for q (16 heads) and k (8 heads). One wave per (t, head).
// Reads fp32 projections, writes bf16 in head-major [h][t][d] layout.
// ---------------------------------------------------------------------------
__global__ __launch_bounds__(256)
void normrope_kernel(const float* __restrict__ qf, const float* __restrict__ kf,
                     const float* __restrict__ cosb, const float* __restrict__ sinb,
                     const float* __restrict__ qw, const float* __restrict__ kw,
                     unsigned short* __restrict__ qb, unsigned short* __restrict__ kb) {
    int gw = (blockIdx.x * 256 + threadIdx.x) >> 5;
    int lane = threadIdx.x & 31;
    int t = gw / (NQH + NKVH);
    int j = gw % (NQH + NKVH);
    if (t >= T_SEQ) return;

    const float* src; const float* w; unsigned short* dst;
    if (j < NQH) {
        src = qf + (size_t)t * (2 * HID) + j * DHEAD;
        w = qw;
        dst = qb + ((size_t)j * T_SEQ + t) * DHEAD;
    } else {
        int hh = j - NQH;
        src = kf + (size_t)t * HID + hh * DHEAD;
        w = kw;
        dst = kb + ((size_t)hh * T_SEQ + t) * DHEAD;
    }

    int p0 = lane * 2;
    float x1a = src[p0],      x1b = src[p0 + 1];
    float x2a = src[p0 + 64], x2b = src[p0 + 65];
    float ss = x1a * x1a + x1b * x1b + x2a * x2a + x2b * x2b;
#pragma unroll
    for (int m = 1; m < 32; m <<= 1) ss += __shfl_xor(ss, m, 32);
    float inv = rsqrtf(ss * (1.0f / DHEAD) + 1e-6f);

    float ca = cosb[t * 64 + p0], cb = cosb[t * 64 + p0 + 1];
    float sa = sinb[t * 64 + p0], sb = sinb[t * 64 + p0 + 1];
    float n1a = x1a * inv * w[p0],      n1b = x1b * inv * w[p0 + 1];
    float n2a = x2a * inv * w[p0 + 64], n2b = x2b * inv * w[p0 + 65];
    dst[p0]      = f2bf(n1a * ca - n2a * sa);
    dst[p0 + 1]  = f2bf(n1b * cb - n2b * sb);
    dst[p0 + 64] = f2bf(n1a * sa + n2a * ca);
    dst[p0 + 65] = f2bf(n1b * sb + n2b * cb);
}

// ---------------------------------------------------------------------------
// V: fp32 [t][h*128+d] -> bf16 transposed [h][d][t]
// ---------------------------------------------------------------------------
__global__ __launch_bounds__(256)
void vtrans_kernel(const float* __restrict__ vf, unsigned short* __restrict__ vtb) {
    int gid = blockIdx.x * 256 + threadIdx.x;
    int t = gid >> 10;
    int c = gid & 1023;
    int h = c >> 7;
    int d = c & 127;
    vtb[((size_t)h * DHEAD + d) * T_SEQ + t] = f2bf(vf[gid]);
}

// ---------------------------------------------------------------------------
// Flash attention, causal, GQA (q head h -> kv head h/2).
// Grid: (32 query blocks, 16 heads). 8 waves x 16 queries = 128-query block.
// K and V^T chunks double-buffered via async global->LDS; online softmax;
// S and PV both via v_wmma_f32_16x16x32_bf16. Output written as bf16 [T,2048].
// ---------------------------------------------------------------------------
__global__ __launch_bounds__(256)
void flash_attn_kernel(const unsigned short* __restrict__ qb,
                       const unsigned short* __restrict__ kb,
                       const unsigned short* __restrict__ vtb,
                       unsigned short* __restrict__ o /* bf16 [T, 2048] */) {
    __shared__ unsigned short k_lds[2][32 * 128];    // [key][d]
    __shared__ unsigned short vt_lds[2][128 * 32];   // [d][key]
    __shared__ unsigned short p_lds[8 * 16 * 32];    // per-wave P transpose

    const int h   = blockIdx.y;
    const int qb0 = blockIdx.x * 128;
    const int kvh = h >> 1;
    const int tid = threadIdx.x;
    const int wave = tid >> 5;
    const int lane = tid & 31;
    const int ln  = lane & 15;
    const int kb8 = (lane >> 4) * 8;
    const int qw0 = qb0 + wave * 16;
    const float scale = 0.08838834764831845f;  // 128^-0.5

    // Q fragments, 4 d-chunks of 32
    BFrag qfr[4];
    {
        const unsigned short* base = qb + ((size_t)h * T_SEQ + qw0 + ln) * DHEAD;
#pragma unroll
        for (int c = 0; c < 4; ++c)
#pragma unroll
            for (int i = 0; i < 8; ++i)
                qfr[c].u[i] = *(const unsigned int*)(base + c * 32 + kpair(i, kb8));
    }

    v8f acc[8];
#pragma unroll
    for (int nt = 0; nt < 8; ++nt) acc[nt] = (v8f)0.0f;
    float mrow[8], lrow[8];
#pragma unroll
    for (int vp = 0; vp < 8; ++vp) { mrow[vp] = -1e30f; lrow[vp] = 0.0f; }

    const int krow = tid >> 3, kseg = (tid & 7) * 16;   // K stage: 16 halfs/thread
    const int vr = tid >> 1, vpart = (tid & 1) * 16;    // Vt stage: 16 halfs/thread
    const unsigned short* gK = kb  + ((size_t)kvh * T_SEQ + krow) * DHEAD + kseg;
    const unsigned short* gV = vtb + ((size_t)kvh * DHEAD + vr) * T_SEQ + vpart;

    const int nch = (qb0 + 128) / 32;
    // prologue: stage chunk 0 into buffer 0
    async_cp32B(gK, &k_lds[0][krow * 128 + kseg]);
    async_cp32B(gV, &vt_lds[0][vr * 32 + vpart]);

    for (int ch = 0; ch < nch; ++ch) {
        const int k0 = ch * 32;
        const int cur = ch & 1;
        if (ch + 1 < nch) {
            __syncthreads();
            async_cp32B(gK + (size_t)(k0 + 32) * DHEAD, &k_lds[cur ^ 1][krow * 128 + kseg]);
            async_cp32B(gV + (k0 + 32), &vt_lds[cur ^ 1][vr * 32 + vpart]);
            wait_async4();
        } else {
            wait_async0();
        }
        __syncthreads();

        const unsigned short* kbuf = k_lds[cur];
        const unsigned short* vbuf = vt_lds[cur];

        // S = Q @ K^T for this wave's 16 queries x 32 keys
        v8f st[2];
        st[0] = (v8f)0.0f; st[1] = (v8f)0.0f;
#pragma unroll
        for (int tn = 0; tn < 2; ++tn) {
            int kr = (tn * 16 + ln) * 128;
#pragma unroll
            for (int c = 0; c < 4; ++c) {
                BFrag bk;
#pragma unroll
                for (int i = 0; i < 8; ++i)
                    bk.u[i] = *(const unsigned int*)&kbuf[kr + c * 32 + kpair(i, kb8)];
                st[tn] = __builtin_amdgcn_wmma_f32_16x16x32_bf16(
                    false, qfr[c].v, false, bk.v, (short)0, st[tn], false, false);
            }
        }

        // scale + causal mask
#pragma unroll
        for (int tn = 0; tn < 2; ++tn) {
            int key = k0 + tn * 16 + ln;
#pragma unroll
            for (int vp = 0; vp < 8; ++vp) {
                int qi = qw0 + vp + kb8;
                float s = st[tn][vp] * scale;
                st[tn][vp] = (key > qi) ? -1e9f : s;
            }
        }

        // row max over 32 keys (reduce across the 16-lane group)
        float rm[8];
#pragma unroll
        for (int vp = 0; vp < 8; ++vp) rm[vp] = fmaxf(st[0][vp], st[1][vp]);
#pragma unroll
        for (int m = 1; m < 16; m <<= 1)
#pragma unroll
            for (int vp = 0; vp < 8; ++vp) rm[vp] = fmaxf(rm[vp], __shfl_xor(rm[vp], m, 16));

        float alpha[8], rs[8];
#pragma unroll
        for (int vp = 0; vp < 8; ++vp) {
            float mnew = fmaxf(mrow[vp], rm[vp]);
            alpha[vp] = __expf(mrow[vp] - mnew);
            mrow[vp] = mnew;
            rs[vp] = 0.0f;
        }
#pragma unroll
        for (int tn = 0; tn < 2; ++tn)
#pragma unroll
            for (int vp = 0; vp < 8; ++vp) {
                float p = __expf(st[tn][vp] - mrow[vp]);
                st[tn][vp] = p;
                rs[vp] += p;
            }
#pragma unroll
        for (int m = 1; m < 16; m <<= 1)
#pragma unroll
            for (int vp = 0; vp < 8; ++vp) rs[vp] += __shfl_xor(rs[vp], m, 16);
#pragma unroll
        for (int vp = 0; vp < 8; ++vp) lrow[vp] = alpha[vp] * lrow[vp] + rs[vp];
#pragma unroll
        for (int nt = 0; nt < 8; ++nt)
#pragma unroll
            for (int vp = 0; vp < 8; ++vp) acc[nt][vp] *= alpha[vp];

        // transpose P through per-wave LDS scratch (C-layout -> A-layout)
        const int woff = wave * 512;
#pragma unroll
        for (int tn = 0; tn < 2; ++tn)
#pragma unroll
            for (int vp = 0; vp < 8; ++vp)
                p_lds[woff + (vp + kb8) * 32 + tn * 16 + ln] = f2bf(st[tn][vp]);
        asm volatile("s_wait_dscnt 0" ::: "memory");

        BFrag pf;
#pragma unroll
        for (int i = 0; i < 8; ++i)
            pf.u[i] = *(const unsigned int*)&p_lds[woff + ln * 32 + kpair(i, kb8)];

        // acc += P @ V  (8 n-tiles covering D=128)
#pragma unroll
        for (int nt = 0; nt < 8; ++nt) {
            BFrag bv;
            int dr = (nt * 16 + ln) * 32;
#pragma unroll
            for (int i = 0; i < 8; ++i)
                bv.u[i] = *(const unsigned int*)&vbuf[dr + kpair(i, kb8)];
            acc[nt] = __builtin_amdgcn_wmma_f32_16x16x32_bf16(
                false, pf.v, false, bv.v, (short)0, acc[nt], false, false);
        }
    }

    // epilogue: o[t][h*128 + d] = bf16(acc / l)  (feeds the Wo GEMM directly)
#pragma unroll
    for (int nt = 0; nt < 8; ++nt)
#pragma unroll
        for (int vp = 0; vp < 8; ++vp) {
            int t = qw0 + vp + kb8;
            o[(size_t)t * (2 * HID) + h * DHEAD + nt * 16 + ln] = f2bf(acc[nt][vp] / lrow[vp]);
        }
}

// ---------------------------------------------------------------------------
extern "C" void kernel_launch(void* const* d_in, const int* in_sizes, int n_in,
                              void* d_out, int out_size, void* d_ws, size_t ws_size,
                              hipStream_t stream) {
    const float* x    = (const float*)d_in[0];
    const float* cosb = (const float*)d_in[1];
    const float* sinb = (const float*)d_in[2];
    const float* Wq   = (const float*)d_in[3];
    const float* Wk   = (const float*)d_in[4];
    const float* Wv   = (const float*)d_in[5];
    const float* Wo   = (const float*)d_in[6];
    const float* qn_w = (const float*)d_in[7];
    const float* kn_w = (const float*)d_in[8];

    const size_t MB = 1048576;
    char* ws = (char*)d_ws;
    float* qf  = (float*)(ws);                   // 32 MB (free after normrope)
    float* kf  = (float*)(ws + 32 * MB);         // 16 MB (reused as attn-out bf16)
    float* vf  = (float*)(ws + 48 * MB);         // 16 MB
    unsigned short* qbuf = (unsigned short*)(ws + 64 * MB);  // 16 MB
    unsigned short* kbuf = (unsigned short*)(ws + 80 * MB);  //  8 MB
    unsigned short* vtb  = (unsigned short*)(ws + 88 * MB);  //  8 MB
    unsigned short* xb   = (unsigned short*)(ws + 96 * MB);  //  8 MB
    unsigned short* Wqt  = (unsigned short*)(ws + 104 * MB); //  4 MB  [2048][1024]
    unsigned short* Wkt  = (unsigned short*)(ws + 108 * MB); //  2 MB  [1024][1024]
    unsigned short* Wvt  = (unsigned short*)(ws + 110 * MB); //  2 MB  [1024][1024]
    unsigned short* Wot  = (unsigned short*)(ws + 112 * MB); //  4 MB  [1024][2048]
    unsigned short* aob  = (unsigned short*)kf;  // attn-out bf16 [T,2048], 16 MB

    dim3 blk(256);
    // one-time conversions: x -> bf16, weights -> bf16 transposed [N][K]
    cvt_bf16_kernel<<<(T_SEQ * HID) / 1024, blk, 0, stream>>>(x, xb);
    cvt_bf16_t_kernel<<<(HID * 2 * HID) / 256, blk, 0, stream>>>(Wq, Wqt, HID, 2 * HID);
    cvt_bf16_t_kernel<<<(HID * HID) / 256, blk, 0, stream>>>(Wk, Wkt, HID, HID);
    cvt_bf16_t_kernel<<<(HID * HID) / 256, blk, 0, stream>>>(Wv, Wvt, HID, HID);
    cvt_bf16_t_kernel<<<(2 * HID * HID) / 256, blk, 0, stream>>>(Wo, Wot, 2 * HID, HID);
    // QKV projections (bf16 x bf16^T -> f32)
    gemm_bf16_128<<<dim3(16, 32), blk, 0, stream>>>(xb, Wqt, qf, T_SEQ, 2 * HID, HID);
    gemm_bf16_128<<<dim3(8, 32),  blk, 0, stream>>>(xb, Wkt, kf, T_SEQ, HID, HID);
    gemm_bf16_128<<<dim3(8, 32),  blk, 0, stream>>>(xb, Wvt, vf, T_SEQ, HID, HID);
    // RMSNorm + RoPE -> bf16 head-major
    normrope_kernel<<<12288, blk, 0, stream>>>(qf, kf, cosb, sinb, qn_w, kn_w, qbuf, kbuf);
    // V -> bf16 [h][d][t]
    vtrans_kernel<<<16384, blk, 0, stream>>>(vf, vtb);
    // flash attention (writes bf16 [T,2048] into the dead kf region)
    flash_attn_kernel<<<dim3(32, 16), blk, 0, stream>>>(qbuf, kbuf, vtb, aob);
    // output projection
    gemm_bf16_128<<<dim3(8, 32), blk, 0, stream>>>(aob, Wot, (float*)d_out, T_SEQ, HID, 2 * HID);
}
